// MoE_21741124452774
// MI455X (gfx1250) — compile-verified, matrix operand order checked by reading, not run on hardware
//
#include <hip/hip_runtime.h>
#include <hip/hip_bf16.h>
#include <math.h>

#define CDIM 1024
#define HDIM 4096
#define EDIM 8
#define NTOK 8192            // B*T = 4*2048
#define HC   128             // H chunk held in LDS
#define NCHUNK (HDIM / HC)   // 32
#define MAXTILES (NTOK / 16 + EDIM)   // 520: worst-case sum of per-expert tiles
#define TT 64                // transpose tile

typedef __attribute__((ext_vector_type(16))) __bf16 v16bf;
typedef __attribute__((ext_vector_type(8)))  float  v8f;

union Frag {
    uint4 q[2];
    unsigned short s[16];
    v16bf v;
};

__device__ __forceinline__ unsigned short f2bf(float f) {
    unsigned u = __builtin_bit_cast(unsigned, f);
    u += 0x7FFFu + ((u >> 16) & 1u);      // round-to-nearest-even
    return (unsigned short)(u >> 16);
}

// ---------------- init: zero per-expert counters ----------------
__global__ void moe_init_kernel(int* cnt) {
    if (threadIdx.x < EDIM) cnt[threadIdx.x] = 0;
}

// ------- weight transpose + fp32->bf16 convert: w[e][K][N] -> wt[e][N][K] -------
__global__ __launch_bounds__(256) void moe_tr_kernel(
        const float* __restrict__ w, unsigned short* __restrict__ wt,
        int K, int N) {
    __shared__ unsigned short tile[TT][TT + 2];   // +2 -> conflict-free transpose
    const int ntn = N / TT;
    const int e  = blockIdx.y;
    const int tk = (blockIdx.x / ntn) * TT;
    const int tn = (blockIdx.x % ntn) * TT;
    const float* we = w + (size_t)e * K * N;
    unsigned short* wte = wt + (size_t)e * K * N;
    const int c  = threadIdx.x & 63;
    const int r0 = threadIdx.x >> 6;              // 0..3
#pragma unroll
    for (int i = 0; i < TT; i += 4) {             // coalesced fp32 reads along N
        int r = r0 + i;
        tile[c][r] = f2bf(we[(size_t)(tk + r) * N + tn + c]);
    }
    __syncthreads();
#pragma unroll
    for (int i = 0; i < TT; i += 4) {             // coalesced bf16 writes along K
        int r = r0 + i;
        wte[(size_t)(tn + r) * K + tk + c] = tile[r][c];
    }
}

// ---------------- router: one wave per token, top-1 argmax ----------------
__global__ __launch_bounds__(256) void moe_router_kernel(
        const float* __restrict__ x, const float* __restrict__ wr,
        const float* __restrict__ br, int* __restrict__ top1) {
    int wave = (int)((blockIdx.x * blockDim.x + threadIdx.x) >> 5);
    int lane = threadIdx.x & 31;
    if (wave >= NTOK) return;
    const float* xr = x + (size_t)wave * CDIM;
    float acc[EDIM];
#pragma unroll
    for (int e = 0; e < EDIM; ++e) acc[e] = 0.0f;
    for (int c = lane; c < CDIM; c += 32) {
        float xv = xr[c];
        const float* w = wr + (size_t)c * EDIM;
#pragma unroll
        for (int e = 0; e < EDIM; ++e) acc[e] += xv * w[e];
    }
#pragma unroll
    for (int e = 0; e < EDIM; ++e)
        for (int off = 16; off > 0; off >>= 1)
            acc[e] += __shfl_xor(acc[e], off, 32);
    if (lane == 0) {
        int best = 0;
        float bv = acc[0] + br[0];
#pragma unroll
        for (int e = 1; e < EDIM; ++e) {
            float v = acc[e] + br[e];
            if (v > bv) { bv = v; best = e; }   // strict > keeps first (jnp.argmax)
        }
        top1[wave] = best;
    }
}

// ---------------- binning: bucket token ids per expert ----------------
__global__ void moe_bin_kernel(const int* __restrict__ top1,
                               int* __restrict__ cnt, int* __restrict__ perm) {
    int n = blockIdx.x * blockDim.x + threadIdx.x;
    if (n >= NTOK) return;
    int e = top1[n];
    int slot = atomicAdd(&cnt[e], 1);
    perm[e * NTOK + slot] = n;
}

// ---------------- fused grouped expert MLP (WMMA bf16) ----------------
// w1t: bf16 [E][H][C]  (transposed w1)   w2t: bf16 [E][C][H]  (transposed w2)
__global__ __launch_bounds__(256) void moe_ffn_kernel(
        const float* __restrict__ x,
        const unsigned short* __restrict__ w1t, const float* __restrict__ b1,
        const unsigned short* __restrict__ w2t, const float* __restrict__ b2,
        const int* __restrict__ cnt, const int* __restrict__ perm,
        float* __restrict__ out) {
    // +8 element row pad -> 4-bank row skew, conflict-free 16B reads
    __shared__ __align__(16) unsigned short x_tile[16][CDIM + 8];
    __shared__ __align__(16) unsigned short h_buf[16][HC + 8];
    __shared__ int s_tok[16];
    __shared__ int s_meta[2];

    // self-assign (expert, tile) from runtime counts
    if (threadIdx.x == 0) {
        int b = blockIdx.x, base = 0, e = -1, tile = 0;
        for (int i = 0; i < EDIM; ++i) {
            int t = (cnt[i] + 15) >> 4;
            if (b < base + t) { e = i; tile = b - base; break; }
            base += t;
        }
        s_meta[0] = e; s_meta[1] = tile;
    }
    __syncthreads();
    const int e = s_meta[0];
    if (e < 0) return;                    // uniform across block
    const int tile = s_meta[1];

    if (threadIdx.x < 16) {
        int idx = tile * 16 + (int)threadIdx.x;
        s_tok[threadIdx.x] = (idx < cnt[e]) ? perm[e * NTOK + idx] : -1;
    }
    __syncthreads();

    // gather x rows -> LDS bf16 (16 threads per row)
    {
        int row = threadIdx.x >> 4;
        int c0  = threadIdx.x & 15;
        int tok = s_tok[row];
        if (tok < 0) tok = s_tok[0];      // pad rows replay row 0, never stored
        const float* xr = x + (size_t)tok * CDIM;
        for (int c = c0; c < CDIM; c += 16)
            x_tile[row][c] = f2bf(xr[c]);
    }
    __syncthreads();

    const int wave  = threadIdx.x >> 5;   // 0..7
    const int lane  = threadIdx.x & 31;
    const int nloc  = lane & 15;
    const int kbase = (lane < 16) ? 0 : 8;   // 16-bit A/B operand lane split
    const int mhi   = (lane >> 4) << 3;      // C/D: rows r (lanes<16) / r+8

    const unsigned short* w1te = w1t + (size_t)e * HDIM * CDIM;
    const unsigned short* w2te = w2t + (size_t)e * CDIM * HDIM;
    const float* b1e = b1 + (size_t)e * HDIM;
    const float* b2e = b2 + (size_t)e * CDIM;

    const v8f vzero = {0.f, 0.f, 0.f, 0.f, 0.f, 0.f, 0.f, 0.f};
    v8f yacc[8];
#pragma unroll
    for (int j = 0; j < 8; ++j) yacc[j] = vzero;

    for (int ch = 0; ch < NCHUNK; ++ch) {
        // -------- stage A: h[16, HC] = gelu(x_tile @ w1[:, chunk]) --------
        const int hc0 = ch * HC + wave * 16;     // this wave's h columns
        v8f acc = vzero;
        // B row for this lane: w1^T[hc0+nloc][*] -- K-contiguous bf16
        const unsigned short* wrow1 = w1te + (size_t)(hc0 + nloc) * CDIM + kbase;
        for (int k0 = 0; k0 < CDIM; k0 += 32) {
            Frag a, b;
            a.q[0] = *(const uint4*)&x_tile[nloc][k0 + kbase];
            a.q[1] = *(const uint4*)&x_tile[nloc][k0 + kbase + 16];
            b.q[0] = *(const uint4*)(wrow1 + k0);
            b.q[1] = *(const uint4*)(wrow1 + k0 + 16);
            __builtin_prefetch(wrow1 + k0 + 128, 0, 1);
            acc = __builtin_amdgcn_wmma_f32_16x16x32_bf16(
                      false, a.v, false, b.v, (short)0, acc, false, false);
        }
        {
            float bias = b1e[hc0 + nloc];
#pragma unroll
            for (int r = 0; r < 8; ++r) {
                float v = acc[r] + bias;
                v = 0.5f * v * (1.0f + erff(v * 0.70710678118654752f));  // exact gelu
                h_buf[r + mhi][wave * 16 + nloc] = f2bf(v);
            }
        }
        __syncthreads();

        // -------- stage B: y += h @ w2[chunk rows, :] --------
#pragma unroll
        for (int j = 0; j < 8; ++j) {
            const int n0 = wave * 128 + j * 16;   // this wave's y columns
            // B row for this lane: w2^T[n0+nloc][*] -- K-contiguous bf16
            const unsigned short* wrow2 =
                w2te + (size_t)(n0 + nloc) * HDIM + ch * HC + kbase;
#pragma unroll
            for (int kk = 0; kk < HC; kk += 32) {
                Frag a, b;
                a.q[0] = *(const uint4*)&h_buf[nloc][kk + kbase];
                a.q[1] = *(const uint4*)&h_buf[nloc][kk + kbase + 16];
                b.q[0] = *(const uint4*)(wrow2 + kk);
                b.q[1] = *(const uint4*)(wrow2 + kk + 16);
                yacc[j] = __builtin_amdgcn_wmma_f32_16x16x32_bf16(
                              false, a.v, false, b.v, (short)0, yacc[j], false, false);
            }
        }
        __syncthreads();   // protect h_buf before next chunk overwrites it
    }

    // -------- epilogue: +b2, scatter rows to their tokens --------
#pragma unroll
    for (int j = 0; j < 8; ++j) {
        const int n0 = wave * 128 + j * 16;
        float b2v = b2e[n0 + nloc];
#pragma unroll
        for (int r = 0; r < 8; ++r) {
            int tok = s_tok[r + mhi];
            if (tok >= 0)
                out[(size_t)tok * CDIM + n0 + nloc] = yacc[j][r] + b2v;
        }
    }
}

extern "C" void kernel_launch(void* const* d_in, const int* in_sizes, int n_in,
                              void* d_out, int out_size, void* d_ws, size_t ws_size,
                              hipStream_t stream) {
    (void)in_sizes; (void)n_in; (void)out_size; (void)ws_size;
    const float* x  = (const float*)d_in[0];
    const float* wr = (const float*)d_in[1];
    const float* br = (const float*)d_in[2];
    const float* w1 = (const float*)d_in[3];
    const float* b1 = (const float*)d_in[4];
    const float* w2 = (const float*)d_in[5];
    const float* b2 = (const float*)d_in[6];
    float* out = (float*)d_out;

    // workspace layout:
    //   [cnt: 8 ints @0] [top1: NTOK ints @256] [perm: E*NTOK ints]
    //   [w1t: bf16 E*H*C @1MB] [w2t: bf16 E*C*H @1MB+64MB]     (~129 MB total)
    int* cnt  = (int*)d_ws;
    int* top1 = (int*)((char*)d_ws + 256);
    int* perm = (int*)((char*)d_ws + 256 + (size_t)NTOK * sizeof(int));
    unsigned short* w1t = (unsigned short*)((char*)d_ws + (size_t)(1 << 20));
    unsigned short* w2t = (unsigned short*)((char*)d_ws + (size_t)(1 << 20)
                                            + (size_t)EDIM * HDIM * CDIM * 2);

    moe_init_kernel<<<1, 64, 0, stream>>>(cnt);
    // pre-pass: transpose + convert weights to bf16 (once per launch)
    moe_tr_kernel<<<dim3((CDIM / TT) * (HDIM / TT), EDIM), 256, 0, stream>>>(
        w1, w1t, CDIM, HDIM);
    moe_tr_kernel<<<dim3((HDIM / TT) * (CDIM / TT), EDIM), 256, 0, stream>>>(
        w2, w2t, HDIM, CDIM);
    moe_router_kernel<<<NTOK / 8, 256, 0, stream>>>(x, wr, br, top1);
    moe_bin_kernel<<<NTOK / 256, 256, 0, stream>>>(top1, cnt, perm);
    moe_ffn_kernel<<<MAXTILES, 256, 0, stream>>>(x, w1t, b1, w2t, b2, cnt, perm, out);
}
